// DecoderRNN_61581241090484
// MI455X (gfx1250) — matrix operand order repaired and checked by reading, not compiled, not run on hardware
//
#include <hip/hip_runtime.h>
#include <hip/hip_bf16.h>

// ---------------------------------------------------------------------------
// DecoderRNN on MI455X (gfx1250): 2-layer RNN, H=4761, B=64, T=128.
// Weights converted once fp32->bf16 (138 MB total -> resident in 192 MB L2),
// GEMMs via V_WMMA_F32_16X16X32_BF16.
// 256-thread blocks = 8 waves computing a 64x32 output slab (4 M-tiles x
// 2 N-tiles, one 16x16 tile per wave) so the WGP cache coalesces the shared
// A/B tile traffic. Dual accumulators break WMMA->WMMA hazard chains.
// ---------------------------------------------------------------------------

#define H_DIM 4761          // hidden size
#define KP    4800          // padded (multiple of 64 -> even K-unroll by 2)
#define BATCH 64
#define T_LEN 128
#define NBLK  (KP / 32)     // 150 blocks: each covers 32 output columns

typedef __attribute__((ext_vector_type(16))) __bf16 v16bf;
typedef __attribute__((ext_vector_type(8)))  __bf16 v8bf;
typedef __attribute__((ext_vector_type(8)))  float  v8f;

// ---- WMMA tile loaders (contiguous k = 16*(lane/16) + s, same in A and B, not the ISA table order) ------

// A-matrix 16x32 (MxK), 16-bit elements, row-major memory (K contiguous).
// lane L: row = m0 + L%16, 16 contiguous K halves at
//   K = kk + (L<16 ? 0 : 16)
__device__ __forceinline__ v16bf load_tileA(const __bf16* __restrict__ base,
                                            int m0, int kk) {
  const int lane  = threadIdx.x & 31;
  const int row   = m0 + (lane & 15);
  const int kbase = kk + ((lane >> 4) << 4);
  const __bf16* p = base + row * KP + kbase;
  union { v16bf v; v8bf h[2]; } u;
  u.h[0] = *(const v8bf*)(p);        // 128-bit load, 16B aligned
  u.h[1] = *(const v8bf*)(p + 8);    // 128-bit load
  return u.v;
}

// B-matrix 32x16 (KxN) where B = W^T, W stored row-major [N][K] (K contig).
// lane L: weight row n = n0 + L%16, 16 contiguous K halves at
//   K = kk + (L<16 ? 0 : 16)
__device__ __forceinline__ v16bf load_tileB(const __bf16* __restrict__ base,
                                            int n0, int kk) {
  const int lane  = threadIdx.x & 31;
  const int row   = n0 + (lane & 15);
  const int kbase = kk + ((lane >> 4) << 4);
  const __bf16* p = base + row * KP + kbase;
  union { v16bf v; v8bf h[2]; } u;
  u.h[0] = *(const v8bf*)(p);
  u.h[1] = *(const v8bf*)(p + 8);
  return u.v;
}

// ---- one-time prep kernels -----------------------------------------------

// fp32 [H_DIM][H_DIM] -> bf16 [KP][KP], zero-padded.
__global__ void __launch_bounds__(256) convert_weight(const float* __restrict__ src,
                                                      __bf16* __restrict__ dst) {
  int idx = blockIdx.x * 256 + threadIdx.x;          // KP*KP = 23,040,000 < 2^31
  if (idx >= KP * KP) return;
  int k = idx % KP;
  int n = idx / KP;
  float v = (n < H_DIM && k < H_DIM) ? src[n * H_DIM + k] : 0.0f;
  dst[idx] = (__bf16)v;
}

// hidden [2][B][H] fp32 -> h0/h1 bf16 [B][KP], zero-padded.
__global__ void __launch_bounds__(256) init_state(const float* __restrict__ hidden,
                                                  __bf16* __restrict__ h0,
                                                  __bf16* __restrict__ h1) {
  int idx = blockIdx.x * 256 + threadIdx.x;
  if (idx >= BATCH * KP) return;
  int k = idx % KP;
  int m = idx / KP;
  float v0 = (k < H_DIM) ? hidden[m * H_DIM + k] : 0.0f;
  float v1 = (k < H_DIM) ? hidden[BATCH * H_DIM + m * H_DIM + k] : 0.0f;
  h0[idx] = (__bf16)v0;
  h1[idx] = (__bf16)v1;
}

__global__ void __launch_bounds__(256) make_bias(const float* __restrict__ b_ih0,
                                                 const float* __restrict__ b_hh0,
                                                 const float* __restrict__ b_ih1,
                                                 const float* __restrict__ b_hh1,
                                                 float* __restrict__ xbias,
                                                 float* __restrict__ bias1) {
  int i = blockIdx.x * 256 + threadIdx.x;
  if (i >= KP) return;
  xbias[i] = (i < H_DIM) ? (b_ih0[i] + b_hh0[i]) : 0.0f;
  bias1[i] = (i < H_DIM) ? (b_ih1[i] + b_hh1[i]) : 0.0f;
}

// ---- per-step GEMM kernels ------------------------------------------------
// Block = 256 threads = 8 waves covering a 64x32 output slab:
//   wave w: m0 = 16*(w & 3), n0 = 32*blockIdx.x + 16*(w >> 2)
// All 8 waves stream the same A/B cache lines -> WGP$ reuse.

// h0n = tanh(xbias + h0 @ W_hh0^T)
__global__ void __launch_bounds__(256) rnn_step_l0(const __bf16* __restrict__ h0_in,
                                                   const __bf16* __restrict__ W0,
                                                   const float*  __restrict__ xbias,
                                                   __bf16* __restrict__ h0_out) {
  const int wave = threadIdx.x >> 5;
  const int lane = threadIdx.x & 31;
  const int m0   = (wave & 3) << 4;
  const int n0   = (blockIdx.x << 5) + ((wave >> 2) << 4);

  v8f acc0 = {};
  v8f acc1 = {};
  for (int kk = 0; kk < KP; kk += 64) {   // K-unroll x2, independent accums
    v16bf a0 = load_tileA(h0_in, m0, kk);
    v16bf b0 = load_tileB(W0,    n0, kk);
    acc0 = __builtin_amdgcn_wmma_f32_16x16x32_bf16(false, a0, false, b0,
                                                   (short)0, acc0, false, false);
    v16bf a1 = load_tileA(h0_in, m0, kk + 32);
    v16bf b1 = load_tileB(W0,    n0, kk + 32);
    acc1 = __builtin_amdgcn_wmma_f32_16x16x32_bf16(false, a1, false, b1,
                                                   (short)0, acc1, false, false);
  }

  const int   n     = n0 + (lane & 15);
  const float bias  = xbias[n];
  const int   mbase = m0 + ((lane >> 4) << 3);   // C/D: M = r + 8*(lane/16)
#pragma unroll
  for (int r = 0; r < 8; ++r) {
    float v = tanhf(acc0[r] + acc1[r] + bias);
    h0_out[(mbase + r) * KP + n] = (__bf16)v;
  }
}

// h1n = tanh(bias1 + h0n @ W_ih1^T + h1 @ W_hh1^T); also writes ys[:, t, :]
__global__ void __launch_bounds__(256) rnn_step_l1(const __bf16* __restrict__ h0n,
                                                   const __bf16* __restrict__ h1_in,
                                                   const __bf16* __restrict__ W1i,
                                                   const __bf16* __restrict__ W1h,
                                                   const float*  __restrict__ bias1,
                                                   __bf16* __restrict__ h1_out,
                                                   float*  __restrict__ out,
                                                   int t) {
  const int wave = threadIdx.x >> 5;
  const int lane = threadIdx.x & 31;
  const int m0   = (wave & 3) << 4;
  const int n0   = (blockIdx.x << 5) + ((wave >> 2) << 4);

  v8f acc0 = {};   // h0n @ W_ih1^T stream
  v8f acc1 = {};   // h1  @ W_hh1^T stream (independent of acc0)
  for (int kk = 0; kk < KP; kk += 32) {
    v16bf a0 = load_tileA(h0n,   m0, kk);
    v16bf b0 = load_tileB(W1i,   n0, kk);
    acc0 = __builtin_amdgcn_wmma_f32_16x16x32_bf16(false, a0, false, b0,
                                                   (short)0, acc0, false, false);
    v16bf a1 = load_tileA(h1_in, m0, kk);
    v16bf b1 = load_tileB(W1h,   n0, kk);
    acc1 = __builtin_amdgcn_wmma_f32_16x16x32_bf16(false, a1, false, b1,
                                                   (short)0, acc1, false, false);
  }

  const int   n     = n0 + (lane & 15);
  const float bias  = bias1[n];
  const int   mbase = m0 + ((lane >> 4) << 3);
#pragma unroll
  for (int r = 0; r < 8; ++r) {
    float v = tanhf(acc0[r] + acc1[r] + bias);
    h1_out[(mbase + r) * KP + n] = (__bf16)v;
    if (n < H_DIM) {
      // out is [B, T, H]
      out[(size_t)(mbase + r) * (size_t)(T_LEN * H_DIM)
          + (size_t)t * H_DIM + n] = v;
    }
  }
}

// ---------------------------------------------------------------------------

extern "C" void kernel_launch(void* const* d_in, const int* in_sizes, int n_in,
                              void* d_out, int out_size, void* d_ws, size_t ws_size,
                              hipStream_t stream) {
  // setup_inputs order:
  // 0 hidden [2,B,H]  1 W_ih0 (unused: input is zeros)  2 W_hh0 [H,H]
  // 3 b_ih0  4 b_hh0  5 W_ih1 [H,H]  6 W_hh1 [H,H]  7 b_ih1  8 b_hh1
  const float* hidden = (const float*)d_in[0];
  const float* W_hh0  = (const float*)d_in[2];
  const float* b_ih0  = (const float*)d_in[3];
  const float* b_hh0  = (const float*)d_in[4];
  const float* W_ih1  = (const float*)d_in[5];
  const float* W_hh1  = (const float*)d_in[6];
  const float* b_ih1  = (const float*)d_in[7];
  const float* b_hh1  = (const float*)d_in[8];
  float* out = (float*)d_out;

  // workspace layout (all offsets multiple of 256 bytes)
  const size_t szW = (size_t)KP * KP * sizeof(__bf16);    // 46,080,000 B
  const size_t szH = (size_t)BATCH * KP * sizeof(__bf16); // 614,400 B
  char* ws = (char*)d_ws;
  __bf16* W0b  = (__bf16*)(ws);
  __bf16* W1ib = (__bf16*)(ws + szW);
  __bf16* W1hb = (__bf16*)(ws + 2 * szW);
  __bf16* h0a  = (__bf16*)(ws + 3 * szW);
  __bf16* h0b  = (__bf16*)(ws + 3 * szW + szH);
  __bf16* h1a  = (__bf16*)(ws + 3 * szW + 2 * szH);
  __bf16* h1b  = (__bf16*)(ws + 3 * szW + 3 * szH);
  float*  xbias = (float*)(ws + 3 * szW + 4 * szH);
  float*  bias1 = (float*)(ws + 3 * szW + 4 * szH + (size_t)KP * sizeof(float));

  // one-time prep: bf16 weight copies (L2-resident afterwards), state, biases
  const int nblkW = (KP * KP + 255) / 256;
  convert_weight<<<nblkW, 256, 0, stream>>>(W_hh0, W0b);
  convert_weight<<<nblkW, 256, 0, stream>>>(W_ih1, W1ib);
  convert_weight<<<nblkW, 256, 0, stream>>>(W_hh1, W1hb);
  init_state<<<(BATCH * KP + 255) / 256, 256, 0, stream>>>(hidden, h0a, h1a);
  make_bias<<<(KP + 255) / 256, 256, 0, stream>>>(b_ih0, b_hh0, b_ih1, b_hh1,
                                                  xbias, bias1);

  // time loop: ping-pong state buffers, two GEMM kernels per step
  const __bf16* h0_cur = h0a;  __bf16* h0_nxt = h0b;
  const __bf16* h1_cur = h1a;  __bf16* h1_nxt = h1b;
  for (int t = 0; t < T_LEN; ++t) {
    rnn_step_l0<<<NBLK, 256, 0, stream>>>(h0_cur, W0b, xbias, h0_nxt);
    rnn_step_l1<<<NBLK, 256, 0, stream>>>(h0_nxt, h1_cur, W1ib, W1hb, bias1,
                                          h1_nxt, out, t);
    const __bf16* tmp0 = h0_cur; h0_cur = h0_nxt; h0_nxt = (__bf16*)tmp0;
    const __bf16* tmp1 = h1_cur; h1_cur = h1_nxt; h1_nxt = (__bf16*)tmp1;
  }
}
